// Pointnet2Backbone_weak_19576460935549
// MI455X (gfx1250) — matrix operand order repaired and checked by reading, not compiled, not run on hardware
//
#include <hip/hip_runtime.h>
#include <hip/hip_bf16.h>

// ---------------------------------------------------------------------------
// PointNet++ backbone for MI455X (gfx1250, wave32).
// All channel-mixing matmuls go through v_wmma_f32_16x16x32_f16 (f32 accum).
// K is padded to multiples of 32 (zero-padded activations + pre-converted
// zero-padded f16 weights) so the WMMA k-loop has no guards or divergence.
// Input-index assumption: d_in[0]=pointcloud, then params in insertion order
// (sa1..sa4, fp1..fp4, fc), each layer flattened (W,g,b).
// ---------------------------------------------------------------------------

typedef __attribute__((ext_vector_type(16))) _Float16 v16h;
typedef __attribute__((ext_vector_type(8)))  float    v8f;

#define BATCH 4
#define NPTS  16384

static inline int roundup32(int x) { return (x + 31) & ~31; }

// =================== Weight convert to padded f16 ==========================
// Wh[Cpad x Kpad] = zero-padded f16 copy of W[Cout x K].
__global__ void weight_cvt_kernel(const float* __restrict__ W,
                                  _Float16* __restrict__ Wh,
                                  int Cout, int K, int Kpad, int total)
{
    const int t = blockIdx.x * blockDim.x + threadIdx.x;
    if (t >= total) return;
    const int n = t / Kpad, k = t % Kpad;
    float v = 0.0f;
    if (n < Cout && k < K) v = W[(size_t)n * K + k];
    Wh[t] = (_Float16)v;
}

// ============================= WMMA GEMM ===================================
// C[M x Cout] = A[M x Kpad] * Wh^T (+ optional bias).
// A: f32, row stride Kpad (zero-padded), M multiple of 64.
// Wh: f16, Cpad x Kpad, zero-padded (rows >= Cout are zero).
// Block = 256 threads = 8 waves; wave grid 4x2 of 16x16 tiles -> 64x32 C tile.
__global__ __launch_bounds__(256) void gemm_wmma_kernel(
    const float* __restrict__ A, const _Float16* __restrict__ Wh,
    const float* __restrict__ bias, float* __restrict__ C,
    int M, int Kpad, int Cout)
{
    const int lane    = threadIdx.x & 31;
    const int wave    = threadIdx.x >> 5;
    const int rowBase = blockIdx.x * 64 + (wave & 3) * 16;
    const int colBase = blockIdx.y * 32 + (wave >> 2) * 16;
    const int m  = rowBase + (lane & 15);
    const int n  = colBase + (lane & 15);
    const bool hi = lane >= 16;

    v8f acc = {};
    const float*    arow = A  + (size_t)m * Kpad + (hi ? 16 : 0);
    const _Float16* wrow = Wh + (size_t)n * Kpad + (hi ? 16 : 0);

    for (int k0 = 0; k0 < Kpad; k0 += 32) {
        // A fragment: lane<16 holds K 0..15, lane>=16 holds K 16..31
        const float4 a0 = *(const float4*)(arow + k0);
        const float4 a1 = *(const float4*)(arow + k0 + 4);
        const float4 a2 = *(const float4*)(arow + k0 + 8);
        const float4 a3 = *(const float4*)(arow + k0 + 12);
        v16h a;
        a[0]  = (_Float16)a0.x; a[1]  = (_Float16)a0.y;
        a[2]  = (_Float16)a0.z; a[3]  = (_Float16)a0.w;
        a[4]  = (_Float16)a1.x; a[5]  = (_Float16)a1.y;
        a[6]  = (_Float16)a1.z; a[7]  = (_Float16)a1.w;
        a[8]  = (_Float16)a2.x; a[9]  = (_Float16)a2.y;
        a[10] = (_Float16)a2.z; a[11] = (_Float16)a2.w;
        a[12] = (_Float16)a3.x; a[13] = (_Float16)a3.y;
        a[14] = (_Float16)a3.z; a[15] = (_Float16)a3.w;
        // B fragment: lane holds column n; K = k0..k0+15 (lo half) / +16 (hi)
        const v16h b = *(const v16h*)(wrow + k0);
        acc = __builtin_amdgcn_wmma_f32_16x16x32_f16(
            false, a, false, b, (short)0, acc, false, false);
    }

    if (n < Cout) {
        const float bv = bias ? bias[n] : 0.0f;
        const int mrow = rowBase + (hi ? 8 : 0);
#pragma unroll
        for (int r = 0; r < 8; ++r)
            C[(size_t)(mrow + r) * Cout + n] = acc[r] + bv;
    }
}

// ===================== Farthest point sampling =============================
__global__ __launch_bounds__(256) void fps_kernel(
    const float* __restrict__ xyz, float* __restrict__ dist,
    int* __restrict__ out, int N, int npoint)
{
    const int b = blockIdx.x;
    const float* px = xyz + (size_t)b * N * 3;
    float* pd = dist + (size_t)b * N;
    int* po = out + (size_t)b * npoint;
    const int tid = threadIdx.x;

    for (int i = tid; i < N; i += 256) pd[i] = 1e10f;

    __shared__ float sbest[256];
    __shared__ int   sbesti[256];
    __shared__ float qx, qy, qz;
    __shared__ int   slast;
    if (tid == 0) { slast = 0; po[0] = 0; }
    __syncthreads();

    for (int it = 1; it < npoint; ++it) {
        if (tid == 0) {
            const int l = slast;
            qx = px[l * 3 + 0]; qy = px[l * 3 + 1]; qz = px[l * 3 + 2];
        }
        __syncthreads();
        const float bx = qx, by = qy, bz = qz;
        float best = -1.0f; int besti = 0;
        for (int i = tid; i < N; i += 256) {
            const float dx = px[i * 3 + 0] - bx;
            const float dy = px[i * 3 + 1] - by;
            const float dz = px[i * 3 + 2] - bz;
            const float dn = fminf(pd[i], dx * dx + dy * dy + dz * dz);
            pd[i] = dn;
            if (dn > best) { best = dn; besti = i; }
        }
        sbest[tid] = best; sbesti[tid] = besti;
        __syncthreads();
        for (int s = 128; s > 0; s >>= 1) {
            if (tid < s) {
                if (sbest[tid + s] > sbest[tid] ||
                    (sbest[tid + s] == sbest[tid] && sbesti[tid + s] < sbesti[tid])) {
                    sbest[tid] = sbest[tid + s]; sbesti[tid] = sbesti[tid + s];
                }
            }
            __syncthreads();
        }
        if (tid == 0) { slast = sbesti[0]; po[it] = sbesti[0]; }
        __syncthreads();
    }
}

__global__ void gather3_kernel(const float* __restrict__ xyz,
                               const int* __restrict__ idx,
                               float* __restrict__ out, int N, int np, int total)
{
    const int t = blockIdx.x * blockDim.x + threadIdx.x;
    if (t >= total) return;
    const int b = t / np;
    const int pi = idx[t];
    const float* s = xyz + ((size_t)b * N + pi) * 3;
    float* d = out + (size_t)t * 3;
    d[0] = s[0]; d[1] = s[1]; d[2] = s[2];
}

// ========================= Ball query ======================================
__global__ __launch_bounds__(256) void ball_query_kernel(
    const float* __restrict__ xyz, const float* __restrict__ new_xyz,
    int* __restrict__ idx, int B, int N, int np, int ns, float r2)
{
    const int gw = (int)((blockIdx.x * blockDim.x + threadIdx.x) >> 5);
    const int lane = threadIdx.x & 31;
    const int total = B * np;
    if (gw >= total) return;
    const int b = gw / np;
    const float* px = xyz + (size_t)b * N * 3;
    const float* q = new_xyz + (size_t)gw * 3;
    const float qx = q[0], qy = q[1], qz = q[2];
    int* po = idx + (size_t)gw * ns;

    int cnt = 0;
    for (int base = 0; base < N && cnt < ns; base += 32) {
        const int i = base + lane;
        bool inb = false;
        if (i < N) {
            const float dx = px[i * 3 + 0] - qx;
            const float dy = px[i * 3 + 1] - qy;
            const float dz = px[i * 3 + 2] - qz;
            inb = (dx * dx + dy * dy + dz * dz) < r2;
        }
        const unsigned mask = (unsigned)__ballot(inb);
        if (inb) {
            const int pos = cnt + __popc(mask & ((1u << lane) - 1u));
            if (pos < ns) po[pos] = i;
        }
        cnt += __popc(mask);
    }
    if (cnt > ns) cnt = ns;
    int padv = 0;
    if (cnt > 0) padv = po[0];          // pad with first hit; 0 if none
    for (int j = cnt + lane; j < ns; j += 32) po[j] = padv;
}

// ========= Grouping + channel concat (xyz | feats), zero-padded K ==========
__global__ void group_concat_kernel(
    const float* __restrict__ xyz, const float* __restrict__ new_xyz,
    const float* __restrict__ feats, const int* __restrict__ idx,
    float* __restrict__ X, int N, int np, int ns, int Cf, int Kpad, float inv_r)
{
    const size_t row = blockIdx.x;              // over B*np*ns
    const size_t pns = (size_t)np * ns;
    const int b = (int)(row / pns);
    const size_t rp = row % pns;
    const int p = (int)(rp / ns);
    const int pi = idx[row];
    const int K = 3 + Cf;
    float* xo = X + row * (size_t)Kpad;
    const float* src = xyz + ((size_t)b * N + pi) * 3;
    const float* q = new_xyz + ((size_t)b * np + p) * 3;
    for (int c = threadIdx.x; c < Kpad; c += blockDim.x) {
        float v = 0.0f;
        if (c < 3)      v = (src[c] - q[c]) * inv_r;
        else if (c < K) v = feats[((size_t)b * N + pi) * Cf + (c - 3)];
        xo[c] = v;
    }
}

// ===================== BatchNorm stats + apply =============================
__global__ __launch_bounds__(256) void bn_stats_kernel(
    const float* __restrict__ Y, float* __restrict__ stats, size_t total, int C)
{
    __shared__ float ssum[256], ssq[256];
    for (int c = threadIdx.x; c < C; c += 256) { ssum[c] = 0.0f; ssq[c] = 0.0f; }
    __syncthreads();
    const size_t stride = (size_t)gridDim.x * 256;
    for (size_t e = (size_t)blockIdx.x * 256 + threadIdx.x; e < total; e += stride) {
        const float v = Y[e];
        const int c = (int)(e % (size_t)C);
        atomicAdd(&ssum[c], v);        // ds_add_f32
        atomicAdd(&ssq[c], v * v);
    }
    __syncthreads();
    for (int c = threadIdx.x; c < C; c += 256) {
        atomicAdd(&stats[c], ssum[c]);
        atomicAdd(&stats[C + c], ssq[c]);
    }
}

__global__ __launch_bounds__(256) void bn_relu_kernel(
    float* __restrict__ Y, const float* __restrict__ stats,
    const float* __restrict__ gamma, const float* __restrict__ beta,
    size_t M, int C)
{
    const size_t total = M * (size_t)C;
    const size_t stride = (size_t)gridDim.x * 256;
    const float invM = 1.0f / (float)M;
    for (size_t e = (size_t)blockIdx.x * 256 + threadIdx.x; e < total; e += stride) {
        const int c = (int)(e % (size_t)C);
        const float mean = stats[c] * invM;
        float var = stats[C + c] * invM - mean * mean;
        var = fmaxf(var, 0.0f);
        const float v = (Y[e] - mean) * rsqrtf(var + 1e-5f) * gamma[c] + beta[c];
        Y[e] = fmaxf(v, 0.0f);
    }
}

// ======================= Max-pool over samples =============================
__global__ void maxpool_kernel(const float* __restrict__ H,
                               float* __restrict__ out, int ns, int C)
{
    const size_t g = blockIdx.x;                // over B*np
    const float* src = H + g * (size_t)ns * C;
    float* dst = out + g * (size_t)C;
    for (int c = threadIdx.x; c < C; c += blockDim.x) {
        float m = src[c];
        for (int s = 1; s < ns; ++s) m = fmaxf(m, src[(size_t)s * C + c]);
        dst[c] = m;
    }
}

// ===================== 3-NN search + interpolation =========================
__global__ void knn3_kernel(const float* __restrict__ unknown,
                            const float* __restrict__ known,
                            int* __restrict__ idx3, float* __restrict__ w3,
                            int B, int nu, int nk)
{
    const int t = blockIdx.x * blockDim.x + threadIdx.x;
    if (t >= B * nu) return;
    const int b = t / nu;
    const float* u = unknown + (size_t)t * 3;
    const float ux = u[0], uy = u[1], uz = u[2];
    const float* kx = known + (size_t)b * nk * 3;
    float d0 = 1e30f, d1 = 1e30f, d2 = 1e30f;
    int i0 = 0, i1 = 0, i2 = 0;
    for (int i = 0; i < nk; ++i) {
        const float dx = kx[i * 3 + 0] - ux;
        const float dy = kx[i * 3 + 1] - uy;
        const float dz = kx[i * 3 + 2] - uz;
        const float d = dx * dx + dy * dy + dz * dz;
        if (d < d0)      { d2 = d1; i2 = i1; d1 = d0; i1 = i0; d0 = d; i0 = i; }
        else if (d < d1) { d2 = d1; i2 = i1; d1 = d; i1 = i; }
        else if (d < d2) { d2 = d; i2 = i; }
    }
    float w0 = 1.0f / (fmaxf(d0, 0.0f) + 1e-8f);
    float w1 = 1.0f / (fmaxf(d1, 0.0f) + 1e-8f);
    float w2 = 1.0f / (fmaxf(d2, 0.0f) + 1e-8f);
    const float s = w0 + w1 + w2;
    idx3[t * 3 + 0] = i0; idx3[t * 3 + 1] = i1; idx3[t * 3 + 2] = i2;
    w3[t * 3 + 0] = w0 / s; w3[t * 3 + 1] = w1 / s; w3[t * 3 + 2] = w2 / s;
}

__global__ void interp_kernel(const float* __restrict__ kf,
                              const int* __restrict__ idx3,
                              const float* __restrict__ w3,
                              float* __restrict__ X,
                              int nu, int nk, int Ck, int K)
{
    const size_t r = blockIdx.x;                // over B*nu
    const int b = (int)(r / (size_t)nu);
    const int* id = idx3 + r * 3;
    const float* w = w3 + r * 3;
    const float* f0 = kf + ((size_t)b * nk + id[0]) * Ck;
    const float* f1 = kf + ((size_t)b * nk + id[1]) * Ck;
    const float* f2 = kf + ((size_t)b * nk + id[2]) * Ck;
    const float w0 = w[0], w1 = w[1], w2 = w[2];
    float* xo = X + r * (size_t)K;
    for (int c = threadIdx.x; c < Ck; c += blockDim.x)
        xo[c] = w0 * f0[c] + w1 * f1[c] + w2 * f2[c];
}

__global__ void copy_skip_kernel(const float* __restrict__ skip,
                                 float* __restrict__ X, int Cs, int K, int off)
{
    const size_t r = blockIdx.x;
    for (int c = threadIdx.x; c < Cs; c += blockDim.x)
        X[r * (size_t)K + off + c] = skip[r * (size_t)Cs + c];
}

// ============================ Host side ====================================
struct LayerP { const float *W, *g, *b; };

static void run_mlp_layer(const float* A, const LayerP& L, float* Y,
                          float* stats, _Float16* Wh,
                          int M, int K, int Kpad, int C, hipStream_t s)
{
    const int Cpad = roundup32(C);
    const int wtot = Cpad * Kpad;
    weight_cvt_kernel<<<(wtot + 255) / 256, 256, 0, s>>>(L.W, Wh, C, K, Kpad, wtot);
    dim3 grid(M / 64, Cpad / 32);
    gemm_wmma_kernel<<<grid, 256, 0, s>>>(A, Wh, nullptr, Y, M, Kpad, C);
    hipMemsetAsync(stats, 0, 2 * (size_t)C * sizeof(float), s);
    bn_stats_kernel<<<512, 256, 0, s>>>(Y, stats, (size_t)M * C, C);
    bn_relu_kernel<<<512, 256, 0, s>>>(Y, stats, L.g, L.b, (size_t)M, C);
}

static void sa_module(const float* xyz_in, int N, const float* feats, int Cf,
                      int np, float radius, int ns,
                      const LayerP* L, const int* chans, int nlayers,
                      float* xyz_out, float* f_out,
                      float* P0, float* P1, _Float16* Wh, int* fidx, int* bidx,
                      float* dists, float* stats, hipStream_t s)
{
    const int tot = BATCH * np;
    fps_kernel<<<BATCH, 256, 0, s>>>(xyz_in, dists, fidx, N, np);
    gather3_kernel<<<(tot + 255) / 256, 256, 0, s>>>(xyz_in, fidx, xyz_out, N, np, tot);
    ball_query_kernel<<<(tot + 7) / 8, 256, 0, s>>>(xyz_in, xyz_out, bidx,
                                                    BATCH, N, np, ns, radius * radius);
    const size_t rows = (size_t)tot * ns;
    const int K0 = 3 + Cf;
    const int Kpad0 = roundup32(K0);
    group_concat_kernel<<<(unsigned)rows, 64, 0, s>>>(xyz_in, xyz_out, feats, bidx,
                                                      P0, N, np, ns, Cf, Kpad0,
                                                      1.0f / radius);
    const float* A = P0;
    int K = K0, Kpad = Kpad0;
    const int M = (int)rows;
    for (int i = 0; i < nlayers; ++i) {
        float* Y = (A == P0) ? P1 : P0;
        run_mlp_layer(A, L[i], Y, stats, Wh, M, K, Kpad, chans[i], s);
        A = Y; K = chans[i]; Kpad = K;   // layer outputs are multiples of 32
    }
    maxpool_kernel<<<tot, 256, 0, s>>>(A, f_out, ns, K);
}

static void fp_module(const float* unknown, int nu, const float* known, int nk,
                      const float* skip, int Cs, const float* kf, int Ck,
                      const LayerP* L, const int* chans, int nlayers,
                      float* g_out, float* P0, float* P1, _Float16* Wh,
                      int* kidx, float* kw, float* stats, hipStream_t s)
{
    const int tot = BATCH * nu;
    knn3_kernel<<<(tot + 255) / 256, 256, 0, s>>>(unknown, known, kidx, kw, BATCH, nu, nk);
    const int K0 = Ck + Cs;              // always a multiple of 32 here
    interp_kernel<<<tot, 256, 0, s>>>(kf, kidx, kw, P0, nu, nk, Ck, K0);
    if (skip) copy_skip_kernel<<<tot, 256, 0, s>>>(skip, P0, Cs, K0, Ck);
    const float* A = P0;
    int K = K0;
    for (int i = 0; i < nlayers; ++i) {
        float* Y = (i == nlayers - 1) ? g_out : ((A == P0) ? P1 : P0);
        run_mlp_layer(A, L[i], Y, stats, Wh, tot, K, K, chans[i], s);
        A = Y; K = chans[i];
    }
}

extern "C" void kernel_launch(void* const* d_in, const int* in_sizes, int n_in,
                              void* d_out, int out_size, void* d_ws, size_t ws_size,
                              hipStream_t stream)
{
    const float* pc = (const float*)d_in[0];   // (4,16384,3)

    int pi = 1;
    auto getLayers = [&](LayerP* dst, int n) {
        for (int i = 0; i < n; ++i) {
            dst[i].W = (const float*)d_in[pi++];
            dst[i].g = (const float*)d_in[pi++];
            dst[i].b = (const float*)d_in[pi++];
        }
    };
    LayerP sa1L[3], sa2L[3], sa3L[3], sa4L[3], fp1L[2], fp2L[2], fp3L[2], fp4L[3];
    getLayers(sa1L, 3); getLayers(sa2L, 3); getLayers(sa3L, 3); getLayers(sa4L, 3);
    getLayers(fp1L, 2); getLayers(fp2L, 2); getLayers(fp3L, 2); getLayers(fp4L, 3);
    const float* fcW1 = (const float*)d_in[pi++];
    const float* fcG1 = (const float*)d_in[pi++];
    const float* fcB1 = (const float*)d_in[pi++];
    const float* fcW2 = (const float*)d_in[pi++];
    const float* fcB2 = (const float*)d_in[pi++];

    // Bump allocator over workspace (~600 MB peak; assumes ws_size suffices).
    char* base = (char*)d_ws;
    size_t cur = 0;
    auto alloc = [&](size_t bytes) -> void* {
        void* p = base + cur;
        cur += (bytes + 255) & ~(size_t)255;
        return p;
    };
    const size_t PSZ = (size_t)524288 * 128;   // largest activation (floats)
    float* P0   = (float*)alloc(PSZ * 4);
    float* P1   = (float*)alloc(PSZ * 4);
    float* xyz1 = (float*)alloc((size_t)BATCH * 2048 * 3 * 4);
    float* xyz2 = (float*)alloc((size_t)BATCH * 1024 * 3 * 4);
    float* xyz3 = (float*)alloc((size_t)BATCH * 512 * 3 * 4);
    float* xyz4 = (float*)alloc((size_t)BATCH * 256 * 3 * 4);
    float* f1   = (float*)alloc((size_t)BATCH * 2048 * 128 * 4);
    float* f2   = (float*)alloc((size_t)BATCH * 1024 * 256 * 4);
    float* f3   = (float*)alloc((size_t)BATCH * 512 * 256 * 4);
    float* f4   = (float*)alloc((size_t)BATCH * 256 * 256 * 4);
    float* g1   = (float*)alloc((size_t)BATCH * 512 * 256 * 4);
    float* g2   = (float*)alloc((size_t)BATCH * 1024 * 256 * 4);
    float* g3   = (float*)alloc((size_t)BATCH * 2048 * 128 * 4);
    float* g4   = (float*)alloc((size_t)BATCH * NPTS * 128 * 4);
    float* dists = (float*)alloc((size_t)BATCH * NPTS * 4);
    int*   fidx  = (int*)  alloc((size_t)BATCH * 2048 * 4);
    int*   bidx  = (int*)  alloc((size_t)BATCH * 2048 * 64 * 4);
    int*   kidx  = (int*)  alloc((size_t)BATCH * NPTS * 3 * 4);
    float* kw    = (float*)alloc((size_t)BATCH * NPTS * 3 * 4);
    float* stats = (float*)alloc(2 * 256 * 4);
    _Float16* Wh = (_Float16*)alloc((size_t)256 * 512 * 2);  // max Cpad*Kpad

    static const int c_sa1[3] = {64, 64, 128};
    static const int c_sa2[3] = {128, 128, 256};
    static const int c_sa3[3] = {128, 128, 256};
    static const int c_sa4[3] = {128, 128, 256};
    static const int c_fp1[2] = {256, 256};
    static const int c_fp2[2] = {256, 256};
    static const int c_fp3[2] = {256, 128};
    static const int c_fp4[3] = {128, 128, 128};

    // ---- Set abstraction chain ----
    sa_module(pc,   NPTS, nullptr, 0,   2048, 0.2f, 64, sa1L, c_sa1, 3, xyz1, f1, P0, P1, Wh, fidx, bidx, dists, stats, stream);
    sa_module(xyz1, 2048, f1,      128, 1024, 0.4f, 32, sa2L, c_sa2, 3, xyz2, f2, P0, P1, Wh, fidx, bidx, dists, stats, stream);
    sa_module(xyz2, 1024, f2,      256, 512,  0.8f, 16, sa3L, c_sa3, 3, xyz3, f3, P0, P1, Wh, fidx, bidx, dists, stats, stream);
    sa_module(xyz3, 512,  f3,      256, 256,  1.2f, 16, sa4L, c_sa4, 3, xyz4, f4, P0, P1, Wh, fidx, bidx, dists, stats, stream);

    // ---- Feature propagation chain ----
    fp_module(xyz3, 512,  xyz4, 256,  f3, 256, f4, 256, fp1L, c_fp1, 2, g1, P0, P1, Wh, kidx, kw, stats, stream);
    fp_module(xyz2, 1024, xyz3, 512,  f2, 256, g1, 256, fp2L, c_fp2, 2, g2, P0, P1, Wh, kidx, kw, stats, stream);
    fp_module(xyz1, 2048, xyz2, 1024, f1, 128, g2, 256, fp3L, c_fp3, 2, g3, P0, P1, Wh, kidx, kw, stats, stream);
    fp_module(pc,   NPTS, xyz1, 2048, nullptr, 0, g3, 128, fp4L, c_fp4, 3, g4, P0, P1, Wh, kidx, kw, stats, stream);

    // ---- Final FC head: gemm -> BN+ReLU -> gemm+bias into d_out ----
    const int M = BATCH * NPTS;   // 65536
    {
        LayerP fc1 = { fcW1, fcG1, fcB1 };
        run_mlp_layer(g4, fc1, P0, stats, Wh, M, 128, 128, 128, stream);

        // last layer: Cout=18 (Cpad=32), bias, no BN
        const int wtot = 32 * 128;
        weight_cvt_kernel<<<(wtot + 255) / 256, 256, 0, stream>>>(fcW2, Wh, 18, 128, 128, wtot);
        dim3 grid2(M / 64, 1);
        gemm_wmma_kernel<<<grid2, 256, 0, stream>>>(P0, Wh, fcB2, (float*)d_out, M, 128, 18);
    }
}
